// ProgressiveFocusedAttention_10385230922119
// MI455X (gfx1250) — matrix operand order repaired and checked by reading, compiled, not run on hardware
//
#include <hip/hip_runtime.h>

// ---------------------------------------------------------------------------
// Progressive Focused (Swin shifted-window) Attention for MI455X / gfx1250.
// bf16 WMMA (v_wmma_f32_16x16x32_bf16) for QKV / QK^T / attn*V / proj GEMMs,
// f32 accumulate. Fully fused per-window; intermediates live in LDS (254 KB).
// x-window gather uses CDNA5 async global->LDS copies (ASYNCcnt).
// ---------------------------------------------------------------------------

typedef unsigned short u16;
typedef __attribute__((ext_vector_type(16))) __bf16 v16bf;
typedef __attribute__((ext_vector_type(8)))  float  v8f;
typedef __attribute__((ext_vector_type(4)))  int    i4v;   // POD 128-bit load

#define BATCH 8
#define HW    128
#define CH    256
#define HEADS 8
#define HD    32
#define WSZ   8
#define L     64          // tokens per window
#define NWIN  2048        // B * NW
#define SHIFT 4
#define SCALEF 0.17677669529663687f   // 32^-0.5

// dynamic-LDS byte offsets (no static LDS in kernel -> dynamic base is 0)
#define OFF_X    0u        // [64][256] bf16  (32 KB)
#define OFF_QKV  32768u    // [64][768] bf16  (96 KB)
#define OFF_VT   131072u   // [256][64] bf16  (32 KB)
#define OFF_S    163840u   // [64][64]  f32   (16 KB)
#define OFF_ABF  180224u   // [64][64]  bf16  ( 8 KB)
#define OFF_O    188416u   // [64][256] f32   (64 KB)  (also x-f32 staging)
#define OFF_REG  253952u   // [64]      int
#define SMEM_BYTES 254208u

// ---- bf16 helpers (round-to-nearest-even) ---------------------------------
__device__ __forceinline__ u16 f2bf(float f) {
  unsigned int u = __float_as_uint(f);
  u += 0x7FFFu + ((u >> 16) & 1u);
  return (u16)(u >> 16);
}
__device__ __forceinline__ float bf2f(u16 h) {
  return __uint_as_float(((unsigned int)h) << 16);
}

union FragU { i4v i4[2]; v16bf v; };

// A fragment: 16x32 bf16, source row-major [M][K] with leading dim ldk.
// Lanes 0-15: row=lane, K = kk..kk+15
// Lanes 16-31: row=lane-16, K = kk+16..kk+31
__device__ __forceinline__ v16bf load_frag_a(const u16* base, int ldk, int lane, int kk) {
  int half = lane >> 4, row = lane & 15;
  const u16* p = base + row * ldk + kk + half * 16;
  FragU u;
  u.i4[0] = *(const i4v*)(p);
  u.i4[1] = *(const i4v*)(p + 8);
  return u.v;
}
// B fragment: 32x16 bf16, source stored [N][K] (i.e. transposed) with ld ldk.
// Lanes 0-15: col=lane, K = kk..kk+15 ; lanes 16-31: col=lane-16, K = kk+16..kk+31
__device__ __forceinline__ v16bf load_frag_b(const u16* base, int ldk, int lane, int kk) {
  int half = lane >> 4, col = lane & 15;
  const u16* p = base + col * ldk + kk + half * 16;
  FragU u;
  u.i4[0] = *(const i4v*)(p);
  u.i4[1] = *(const i4v*)(p + 8);
  return u.v;
}

__device__ __forceinline__ v8f wmma_bf16(v16bf a, v16bf b, v8f c) {
  return __builtin_amdgcn_wmma_f32_16x16x32_bf16(false, a, false, b, (short)0, c, false, false);
}

// CDNA5 async global->LDS copy, 16 bytes per lane (GV mode), ASYNCcnt-tracked.
__device__ __forceinline__ void async_copy_b128(unsigned lds_off, const void* gaddr) {
  asm volatile("global_load_async_to_lds_b128 %0, %1, off"
               :: "v"(lds_off), "v"(gaddr) : "memory");
}
__device__ __forceinline__ void wait_asynccnt0() {
  asm volatile("s_wait_asynccnt 0" ::: "memory");
}

// ---- prep: transpose+convert weights into workspace (bf16, [N][K]) --------
__global__ __launch_bounds__(256) void pfa_prep(const float* __restrict__ qkv_w,
                                                const float* __restrict__ proj_w,
                                                u16* __restrict__ wQT,
                                                u16* __restrict__ wPT) {
  int rrow = blockIdx.x, tid = threadIdx.x;
  if (rrow < 768) {
    // qkv_w is (K=256, N=768) -> wQT[N][K]
    wQT[rrow * 256 + tid] = f2bf(qkv_w[tid * 768 + rrow]);
  } else {
    int r2 = rrow - 768;   // proj_w is (K=256, N=256) -> wPT[N][K]
    wPT[r2 * 256 + tid] = f2bf(proj_w[tid * 256 + r2]);
  }
}

// ---- fused per-window kernel ----------------------------------------------
__global__ __launch_bounds__(256) void pfa_main(
    const float* __restrict__ x, const float* __restrict__ prev,
    const float* __restrict__ qkv_b, const float* __restrict__ proj_b,
    const float* __restrict__ lepe_w, const float* __restrict__ lepe_b,
    const u16* __restrict__ wQT, const u16* __restrict__ wPT,
    float* __restrict__ out, float* __restrict__ attn_out)
{
  extern __shared__ unsigned char smem[];
  u16*   sX   = (u16*)  (smem + OFF_X);    // [64][256] window tokens (bf16), reused for proj-A
  u16*   sQKV = (u16*)  (smem + OFF_QKV);  // [64][768] q|k|v (bf16)
  u16*   sVT  = (u16*)  (smem + OFF_VT);   // [256][64] v transposed (bf16)
  float* sS   = (float*)(smem + OFF_S);    // [64][64]  scores (f32)
  u16*   sA   = (u16*)  (smem + OFF_ABF);  // [64][64]  attn (bf16)
  float* sO   = (float*)(smem + OFF_O);    // [64][256] out accumulator / x staging (f32)
  int*   sReg = (int*)  (smem + OFF_REG);  // [64]      swin mask region ids

  const int tid  = threadIdx.x;
  const int lane = tid & 31, wid = tid >> 5;
  const int win  = blockIdx.x;                    // 0..2047
  const int b    = win >> 8, widx = win & 255;
  const int wi   = widx >> 4, wj = widx & 15;

  // ---- async gather of the cyclically-shifted window into LDS (f32) -------
  // xs[h] = x[(h+SHIFT)%HW]; each lane moves 16B, 16 iterations => 64 KB.
#pragma unroll 4
  for (int i = 0; i < 16; ++i) {
    int e = (i * 256 + tid) * 4;        // float index within [64][256]
    int t = e >> 8, c = e & 255;
    int r = t >> 3, cc = t & 7;
    int hh = (wi * WSZ + r + SHIFT) & (HW - 1);
    int ww = (wj * WSZ + cc + SHIFT) & (HW - 1);
    const float* gp = x + ((((size_t)b * HW + hh) * HW) + ww) * CH + c;
    async_copy_b128(OFF_O + (unsigned)e * 4u, gp);
  }

  // region ids for the shifted-window attention mask (built on unshifted img)
  if (tid < 64) {
    int r = tid >> 3, cc = tid & 7;
    int hc = wi * WSZ + r, wc = wj * WSZ + cc;
    int rh = hc < (HW - WSZ) ? 0 : (hc < (HW - SHIFT) ? 1 : 2);
    int rw = wc < (HW - WSZ) ? 0 : (wc < (HW - SHIFT) ? 1 : 2);
    sReg[tid] = rh * 3 + rw;
  }

  wait_asynccnt0();
  __syncthreads();

  // convert staged f32 window -> bf16 A-operand layout
  for (int idx = tid; idx < L * CH; idx += 256) sX[idx] = f2bf(sO[idx]);
  __syncthreads();

  // ---- QKV GEMM: [64x256] @ [256x768] ; wave tiles 16x16, K-steps of 32 ---
  for (int nt = wid * 6; nt < wid * 6 + 6; ++nt) {
    int nbase = nt * 16;
    for (int mt = 0; mt < 4; ++mt) {
      v8f acc = {0.f, 0.f, 0.f, 0.f, 0.f, 0.f, 0.f, 0.f};
      for (int kk = 0; kk < 256; kk += 32) {
        v16bf af = load_frag_a(sX + mt * 16 * 256, 256, lane, kk);
        v16bf bf = load_frag_b(wQT + (size_t)nbase * 256, 256, lane, kk);
        acc = wmma_bf16(af, bf, acc);
      }
      int half = lane >> 4, col = lane & 15;
      int n = nbase + col;
      float bias = qkv_b[n];
#pragma unroll
      for (int i = 0; i < 8; ++i) {
        int m = mt * 16 + half * 8 + i;
        sQKV[m * 768 + n] = f2bf(acc[i] + bias);
      }
    }
  }

  // warm L2/L0 with proj weights for the epilogue GEMM (global_prefetch_b8)
  for (int off = tid; off < 1024; off += 256)
    __builtin_prefetch(wPT + off * 64, 0, 1);

  __syncthreads();

  // ---- v transposed [C][L] for attn@v B-fragments and LePE ----------------
  for (int idx = tid; idx < L * CH; idx += 256) {
    int c = idx >> 6, t = idx & 63;
    sVT[c * 64 + t] = sQKV[t * 768 + 512 + c];
  }
  __syncthreads();

  // ---- LePE depthwise 3x3 (per window, zero-padded) into sO ---------------
  for (int idx = tid; idx < L * CH; idx += 256) {
    int t = idx >> 8, c = idx & 255;
    int r = t >> 3, cc = t & 7;
    float acc = lepe_b[c];
#pragma unroll
    for (int dr = -1; dr <= 1; ++dr)
#pragma unroll
      for (int dc = -1; dc <= 1; ++dc) {
        int rr = r + dr, c2 = cc + dc;
        if (rr >= 0 && rr < 8 && c2 >= 0 && c2 < 8)
          acc += bf2f(sVT[c * 64 + rr * 8 + c2]) *
                 lepe_w[((dr + 1) * 3 + (dc + 1)) * CH + c];
      }
    sO[t * CH + c] = acc;
  }
  __syncthreads();

  // ---- per-head attention -------------------------------------------------
  for (int h = 0; h < HEADS; ++h) {
    // scores = q @ k^T : M=64,N=64,K=32 -> 16 tiles, 2 per wave
    for (int tt = wid * 2; tt < wid * 2 + 2; ++tt) {
      int mt = tt >> 2, nt = tt & 3;
      v8f acc = {0.f, 0.f, 0.f, 0.f, 0.f, 0.f, 0.f, 0.f};
      v16bf af = load_frag_a(sQKV + mt * 16 * 768 + h * 32, 768, lane, 0);        // q
      v16bf bf = load_frag_b(sQKV + nt * 16 * 768 + 256 + h * 32, 768, lane, 0);  // k rows = [N][K]
      acc = wmma_bf16(af, bf, acc);
      int half = lane >> 4, col = lane & 15;
      int kc = nt * 16 + col;
      int regK = sReg[kc];
#pragma unroll
      for (int i = 0; i < 8; ++i) {
        int qr = mt * 16 + half * 8 + i;
        float s = acc[i] * SCALEF;
        if (sReg[qr] != regK) s += -100.0f;
        float p = prev[(((size_t)win * HEADS + h) * 64 + qr) * 64 + kc];
        sS[qr * 64 + kc] = s * p;      // progressive focusing: (scores+mask)*prev
      }
    }
    __syncthreads();

    // softmax over rows of 64 (4 threads per row, wave32 shfl reduction)
    {
      int r = tid >> 2, qd = tid & 3;
      float* row = sS + r * 64 + qd * 16;
      float vals[16];
      float mx = -3.0e38f;
#pragma unroll
      for (int j = 0; j < 16; ++j) { vals[j] = row[j]; mx = fmaxf(mx, vals[j]); }
      mx = fmaxf(mx, __shfl_xor(mx, 1, 32));
      mx = fmaxf(mx, __shfl_xor(mx, 2, 32));
      float sum = 0.f;
#pragma unroll
      for (int j = 0; j < 16; ++j) { vals[j] = __expf(vals[j] - mx); sum += vals[j]; }
      sum += __shfl_xor(sum, 1, 32);
      sum += __shfl_xor(sum, 2, 32);
      float inv = 1.0f / sum;
      float* gattn = attn_out + (((size_t)win * HEADS + h) * 64 + r) * 64 + qd * 16;
      u16* abf = sA + r * 64 + qd * 16;
#pragma unroll
      for (int j = 0; j < 16; ++j) {
        float a = vals[j] * inv;
        gattn[j] = a;                  // second output: attention tensor
        abf[j] = f2bf(a);
      }
    }
    __syncthreads();

    // out_h = attn @ v : M=64,N=32,K=64 -> 8 tiles, 1 per wave
    {
      int mt = wid >> 1, nt = wid & 1;
      v8f acc = {0.f, 0.f, 0.f, 0.f, 0.f, 0.f, 0.f, 0.f};
      for (int kk = 0; kk < 64; kk += 32) {
        v16bf af = load_frag_a(sA + mt * 16 * 64, 64, lane, kk);
        v16bf bf = load_frag_b(sVT + (h * 32 + nt * 16) * 64, 64, lane, kk);
        acc = wmma_bf16(af, bf, acc);
      }
      int half = lane >> 4, col = lane & 15;
      int c = h * 32 + nt * 16 + col;
#pragma unroll
      for (int i = 0; i < 8; ++i) {
        int m = mt * 16 + half * 8 + i;
        sO[m * CH + c] += acc[i];
      }
    }
    __syncthreads();
  }

  // ---- proj GEMM: [64x256] @ [256x256] + bias, scatter with unshift -------
  for (int idx = tid; idx < L * CH; idx += 256) sX[idx] = f2bf(sO[idx]);
  __syncthreads();

  for (int nt2 = 0; nt2 < 2; ++nt2) {
    int nt = wid * 2 + nt2, nbase = nt * 16;
    for (int mt = 0; mt < 4; ++mt) {
      v8f acc = {0.f, 0.f, 0.f, 0.f, 0.f, 0.f, 0.f, 0.f};
      for (int kk = 0; kk < 256; kk += 32) {
        v16bf af = load_frag_a(sX + mt * 16 * 256, 256, lane, kk);
        v16bf bf = load_frag_b(wPT + (size_t)nbase * 256, 256, lane, kk);
        acc = wmma_bf16(af, bf, acc);
      }
      int half = lane >> 4, col = lane & 15;
      int c = nbase + col;
      float bias = proj_b[c];
#pragma unroll
      for (int i = 0; i < 8; ++i) {
        int t = mt * 16 + half * 8 + i;
        int r = t >> 3, cc = t & 7;
        int hh = (wi * WSZ + r + SHIFT) & (HW - 1);   // window reverse + roll(+SHIFT)
        int ww = (wj * WSZ + cc + SHIFT) & (HW - 1);
        out[((((size_t)b * HW + hh) * HW) + ww) * CH + c] = acc[i] + bias;
      }
    }
  }
}

extern "C" void kernel_launch(void* const* d_in, const int* in_sizes, int n_in,
                              void* d_out, int out_size, void* d_ws, size_t ws_size,
                              hipStream_t stream) {
  (void)in_sizes; (void)n_in; (void)out_size; (void)ws_size;
  const float* x      = (const float*)d_in[0];
  const float* prev   = (const float*)d_in[1];
  const float* qkv_w  = (const float*)d_in[2];
  const float* qkv_b  = (const float*)d_in[3];
  const float* proj_w = (const float*)d_in[4];
  const float* proj_b = (const float*)d_in[5];
  const float* lepe_w = (const float*)d_in[6];
  const float* lepe_b = (const float*)d_in[7];

  float* out  = (float*)d_out;
  float* attn = out + (size_t)BATCH * HW * HW * CH;   // second tuple element

  u16* wQT = (u16*)d_ws;                // [768][256] bf16 (384 KB)
  u16* wPT = wQT + 768 * 256;           // [256][256] bf16 (128 KB)

  pfa_prep<<<1024, 256, 0, stream>>>(qkv_w, proj_w, wQT, wPT);

  pfa_main<<<NWIN, 256, SMEM_BYTES, stream>>>(x, prev, qkv_b, proj_b,
                                              lepe_w, lepe_b, wQT, wPT, out, attn);
}
